// GATModel_55009941127942
// MI455X (gfx1250) — matrix operand order repaired and checked by reading, not compiled, not run on hardware
//
#include <hip/hip_runtime.h>
#include <hip/hip_bf16.h>

typedef __attribute__((ext_vector_type(16))) _Float16 v16h;
typedef __attribute__((ext_vector_type(8)))  float    v8f;

#define NEG_SLOPE 0.2f

// ------------------------------------------------------------------ fill ----
__global__ void fill_kernel(float* __restrict__ p, float v, int n) {
  int i = blockIdx.x * blockDim.x + threadIdx.x;
  if (i < n) p[i] = v;
}

// -------------------------------------------------- WMMA GEMM (fp32->f16) ----
// C[M,Nout] = A[M,K] @ W[K,Nout].  One wave computes a 16 x (16*NT) strip:
// A operand converted once per K-step and reused across NT WMMAs.
// Requires M % 16 == 0 (true for all launches here: M = 50000).
template <int K, int Nout, int NT>
__global__ void gemm_wmma_f16(const float* __restrict__ A,
                              const float* __restrict__ W,
                              float* __restrict__ C, int M) {
  const int wave = (blockIdx.x * blockDim.x + threadIdx.x) >> 5;
  const int lane = threadIdx.x & 31;
  const int ngrp   = Nout / (16 * NT);
  const int mtiles = M >> 4;
  if (wave >= mtiles * ngrp) return;          // wave-uniform: EXEC stays full
  const int tm = wave / ngrp;
  const int tg = wave - tm * ngrp;
  const int hb = lane >> 4;                   // lane-half (0: lanes 0-15)
  const int m  = tm * 16 + (lane & 15);

  v8f acc[NT];
#pragma unroll
  for (int t = 0; t < NT; ++t) acc[t] = (v8f){0.f,0.f,0.f,0.f,0.f,0.f,0.f,0.f};

#pragma unroll
  for (int k0 = 0; k0 < K; k0 += 32) {
    // A-operand: lane = row m; K runs [k0+8h .. +7] and [k0+16+8h .. +7]
    const float* ap = A + (size_t)m * K + k0 + hb * 8;
    v16h av;
#pragma unroll
    for (int i = 0; i < 8; ++i) {
      av[i]     = (_Float16)ap[i];
      av[i + 8] = (_Float16)ap[16 + i];
    }
    // B-operand: lane = N column (lane&15); K runs [k0+8h .. +7] and [k0+16+8h .. +7]
    const float* bp = W + (size_t)(k0 + hb * 8) * Nout + tg * (16 * NT) + (lane & 15);
#pragma unroll
    for (int t = 0; t < NT; ++t) {
      v16h bv;
#pragma unroll
      for (int i = 0; i < 8; ++i) {
        bv[i]     = (_Float16)bp[(size_t)i * Nout + t * 16];
        bv[i + 8] = (_Float16)bp[(size_t)(16 + i) * Nout + t * 16];
      }
      acc[t] = __builtin_amdgcn_wmma_f32_16x16x32_f16(
          /*neg_a=*/false, av, /*neg_b=*/false, bv,
          /*c_mod=*/(short)0, acc[t], /*reuse_a=*/false, /*reuse_b=*/false);
    }
  }

  // D: acc[t][j] -> row tm*16 + hb*8 + j, col tg*(16*NT) + t*16 + (lane&15)
  const int row0 = tm * 16 + hb * 8;
#pragma unroll
  for (int t = 0; t < NT; ++t) {
    float* cp = C + (size_t)row0 * Nout + tg * (16 * NT) + t * 16 + (lane & 15);
#pragma unroll
    for (int j = 0; j < 8; ++j) cp[(size_t)j * Nout] = acc[t][j];
  }
}

// ---------------------------------------------- ordered-float max helpers ----
__device__ __forceinline__ unsigned f32_order(float x) {
  unsigned b = __float_as_uint(x);
  return (b & 0x80000000u) ? ~b : (b | 0x80000000u);
}
__device__ __forceinline__ float f32_unorder(unsigned u) {
  return __uint_as_float((u & 0x80000000u) ? (u & 0x7FFFFFFFu) : ~u);
}

// -------------------------------------- pass A: logits + segment max (dst) ---
// One wave per edge; F/32 channels per lane; heads partition the lanes.
template <int F, int HEADS>
__global__ void edge_logits_max(const long long* __restrict__ esrc,
                                const long long* __restrict__ edst,
                                const float* __restrict__ xl,
                                const float* __restrict__ xr,
                                const float* __restrict__ att,   // [HEADS][F/HEADS]
                                float* __restrict__ logits,      // [EP][HEADS]
                                unsigned* __restrict__ mx,       // [N][HEADS]
                                int E, int EP) {
  const int CPL = F / 32;
  const int C   = F / HEADS;
  const int LPH = 32 / HEADS;
  const int e = (blockIdx.x * blockDim.x + threadIdx.x) >> 5;
  if (e >= EP) return;
  const int lane = threadIdx.x & 31;
  long long s, d;
  if (e < E) { s = esrc[e]; d = edst[e]; } else { s = e - E; d = s; }

  const int c0   = lane * CPL;
  const int head = c0 / C;
  const float* pl = xl + (size_t)s * F + c0;
  const float* pr = xr + (size_t)d * F + c0;
  float part = 0.f;
#pragma unroll
  for (int j = 0; j < CPL; ++j) {
    float v = pl[j] + pr[j];
    v = (v > 0.f) ? v : NEG_SLOPE * v;
    part += v * att[head * C + (c0 % C) + j];
  }
  for (int off = LPH >> 1; off > 0; off >>= 1)
    part += __shfl_xor(part, off, 32);
  if ((lane & (LPH - 1)) == 0) {
    logits[(size_t)e * HEADS + head] = part;
    atomicMax(mx + (size_t)d * HEADS + head, f32_order(part));
  }
}

// ------------------------------- pass B: exp(logit - max), segment sum ------
template <int HEADS>
__global__ void edge_softnorm(const long long* __restrict__ edst,
                              float* __restrict__ logits,        // in: logit, out: exp
                              const unsigned* __restrict__ mx,
                              float* __restrict__ den,
                              int E, int EP) {
  const int e = blockIdx.x * blockDim.x + threadIdx.x;
  if (e >= EP) return;
  const long long d = (e < E) ? edst[e] : (long long)(e - E);
#pragma unroll
  for (int h = 0; h < HEADS; ++h) {
    const float m  = f32_unorder(mx[(size_t)d * HEADS + h]);
    const float ex = __expf(logits[(size_t)e * HEADS + h] - m);
    logits[(size_t)e * HEADS + h] = ex;
    atomicAdd(den + (size_t)d * HEADS + h, ex);
  }
}

// ----------------------- pass C: acc[dst] += alpha * xl[src] (scatter-add) ---
template <int F, int HEADS>
__global__ void edge_aggregate(const long long* __restrict__ esrc,
                               const long long* __restrict__ edst,
                               const float* __restrict__ xl,
                               const float* __restrict__ exs,    // [EP][HEADS]
                               const float* __restrict__ den,    // [N][HEADS]
                               float* __restrict__ acc,          // [N][F]
                               int E, int EP) {
  const int CPL = F / 32;
  const int C   = F / HEADS;
  const int e = (blockIdx.x * blockDim.x + threadIdx.x) >> 5;
  if (e >= EP) return;
  const int lane = threadIdx.x & 31;
  long long s, d;
  if (e < E) { s = esrc[e]; d = edst[e]; } else { s = e - E; d = s; }
  const int c0   = lane * CPL;
  const int head = c0 / C;
  const float alpha = exs[(size_t)e * HEADS + head] /
                      den[(size_t)d * HEADS + head];
  const float* pl = xl + (size_t)s * F + c0;
  float*       pa = acc + (size_t)d * F + c0;
#pragma unroll
  for (int j = 0; j < CPL; ++j) atomicAdd(pa + j, alpha * pl[j]);
}

// ----------------------------------------------------- bias (+ELU) ----------
__global__ void finalize_kernel(const float* __restrict__ acc,
                                const float* __restrict__ bias,
                                float* __restrict__ out,
                                int total, int Fw, int do_elu) {
  const int i = blockIdx.x * blockDim.x + threadIdx.x;
  if (i >= total) return;
  float v = acc[i] + bias[i % Fw];
  if (do_elu) v = (v > 0.f) ? v : (__expf(v) - 1.f);
  out[i] = v;
}

// ---------------------------------------------------------------- host ------
static inline int cdiv(int a, int b) { return (a + b - 1) / b; }

template <int K, int Nout>
static void launch_gemm(const float* A, const float* W, float* C,
                        int M, hipStream_t s) {
  constexpr int NT = 4;                       // 16x64 strip per wave
  const int waves = (M / 16) * (Nout / (16 * NT));
  gemm_wmma_f16<K, Nout, NT><<<cdiv(waves, 8), 256, 0, s>>>(A, W, C, M);
}

template <int F, int HEADS>
static void attention_passes(const long long* esrc, const long long* edst,
                             const float* xl, const float* xr, const float* att,
                             float* logits, unsigned* mx, float* den, float* acc,
                             int N, int E, int EP, hipStream_t s) {
  fill_kernel<<<cdiv(N * HEADS, 256), 256, 0, s>>>((float*)mx, 0.f, N * HEADS);
  fill_kernel<<<cdiv(N * HEADS, 256), 256, 0, s>>>(den, 0.f, N * HEADS);
  fill_kernel<<<cdiv(N * F, 256), 256, 0, s>>>(acc, 0.f, N * F);
  edge_logits_max<F, HEADS><<<cdiv(EP * 32, 256), 256, 0, s>>>(
      esrc, edst, xl, xr, att, logits, mx, E, EP);
  edge_softnorm<HEADS><<<cdiv(EP, 256), 256, 0, s>>>(edst, logits, mx, den, E, EP);
  edge_aggregate<F, HEADS><<<cdiv(EP * 32, 256), 256, 0, s>>>(
      esrc, edst, xl, logits, den, acc, E, EP);
}

extern "C" void kernel_launch(void* const* d_in, const int* in_sizes, int n_in,
                              void* d_out, int out_size, void* d_ws, size_t ws_size,
                              hipStream_t stream) {
  (void)n_in; (void)out_size; (void)ws_size;
  const float*     x    = (const float*)d_in[0];
  const long long* eidx = (const long long*)d_in[1];   // int64 [2, E]
  const float* Wl1 = (const float*)d_in[2];
  const float* Wr1 = (const float*)d_in[3];
  const float* a1  = (const float*)d_in[4];
  const float* b1  = (const float*)d_in[5];
  const float* Wl2 = (const float*)d_in[6];
  const float* Wr2 = (const float*)d_in[7];
  const float* a2  = (const float*)d_in[8];
  const float* b2  = (const float*)d_in[9];
  const float* Wl3 = (const float*)d_in[10];
  const float* Wr3 = (const float*)d_in[11];
  const float* a3  = (const float*)d_in[12];
  const float* b3  = (const float*)d_in[13];
  float* out = (float*)d_out;

  const int N  = in_sizes[0] / 128;   // 50000 (multiple of 16)
  const int E  = in_sizes[1] / 2;     // 800000
  const int EP = E + N;               // + self loops

  // workspace layout (floats)
  float* ws = (float*)d_ws;
  size_t o = 0;
  float*    xl     = ws + o; o += (size_t)N * 128;
  float*    xr     = ws + o; o += (size_t)N * 128;
  float*    hbuf   = ws + o; o += (size_t)N * 128;
  float*    acc    = ws + o; o += (size_t)N * 128;
  float*    logits = ws + o; o += (size_t)EP * 4;
  unsigned* mx     = (unsigned*)(ws + o); o += (size_t)N * 4;
  float*    den    = ws + o; o += (size_t)N * 4;

  const long long* esrc = eidx;
  const long long* edst = eidx + E;

  // ---------------- layer 1: 128 -> 4 heads x 32, concat, ELU ----------------
  launch_gemm<128, 128>(x, Wl1, xl, N, stream);
  launch_gemm<128, 128>(x, Wr1, xr, N, stream);
  attention_passes<128, 4>(esrc, edst, xl, xr, a1, logits, mx, den, acc,
                           N, E, EP, stream);
  finalize_kernel<<<cdiv(N * 128, 256), 256, 0, stream>>>(acc, b1, hbuf,
                                                          N * 128, 128, 1);

  // ---------------- layer 2: 128 -> 4 heads x 32, concat, ELU ----------------
  launch_gemm<128, 128>(hbuf, Wl2, xl, N, stream);
  launch_gemm<128, 128>(hbuf, Wr2, xr, N, stream);
  attention_passes<128, 4>(esrc, edst, xl, xr, a2, logits, mx, den, acc,
                           N, E, EP, stream);
  finalize_kernel<<<cdiv(N * 128, 256), 256, 0, stream>>>(acc, b2, hbuf,
                                                          N * 128, 128, 1);

  // ---------------- layer 3: 128 -> 64, single head, mean(=identity) ---------
  launch_gemm<128, 64>(hbuf, Wl3, xl, N, stream);
  launch_gemm<128, 64>(hbuf, Wr3, xr, N, stream);
  attention_passes<64, 1>(esrc, edst, xl, xr, a3, logits, mx, den, acc,
                          N, E, EP, stream);
  finalize_kernel<<<cdiv(N * 64, 256), 256, 0, stream>>>(acc, b3, out,
                                                         N * 64, 64, 0);
}
